// BagAttentionNet_89532888252840
// MI455X (gfx1250) — matrix operand order repaired and checked by reading, not compiled, not run on hardware
//
#include <hip/hip_runtime.h>
#include <hip/hip_bf16.h>
#include <cstdint>

typedef __attribute__((ext_vector_type(16))) _Float16 v16h;
typedef __attribute__((ext_vector_type(8)))  _Float16 v8h;
typedef __attribute__((ext_vector_type(8)))  float    v8f;

#define TAU 0.95f
#define KEEP 20   // C=64, k = int(64*0.7)=44 dropped -> 20 kept

// ---------------------------------------------------------------------------
// CDNA5 async direct global->LDS copy (16B per lane), tracked by ASYNCcnt.
// ---------------------------------------------------------------------------
__device__ __forceinline__ void async_copy_b128(unsigned lds_addr, const void* gaddr) {
    asm volatile("global_load_async_to_lds_b128 %0, %1, off"
                 :: "v"(lds_addr), "v"(gaddr) : "memory");
}
__device__ __forceinline__ void wait_async0() {
    asm volatile("s_wait_asynccnt 0x0" ::: "memory");
}
__device__ __forceinline__ unsigned lds_off(const void* p) {
    // LDS flat addresses: low 32 bits are the LDS byte address
    return (unsigned)(uintptr_t)p;
}

// ---------------------------------------------------------------------------
// Weight transpose + fp32 -> f16 convert:  W (K x N, row major) -> Wt (N x K)
// ---------------------------------------------------------------------------
__global__ __launch_bounds__(256) void convert_wt(const float* __restrict__ W,
                                                  _Float16* __restrict__ Wt,
                                                  int K, int N) {
    int idx = blockIdx.x * 256 + threadIdx.x;
    if (idx < K * N) {
        int n = idx / K;
        int k = idx - n * K;
        Wt[idx] = (_Float16)W[(size_t)k * N + n];
    }
}

// ---------------------------------------------------------------------------
// WMMA GEMM:  C = act(A(MxK) @ B(KxN) + bias),  B given transposed (N x K) f16
// Block tile 128 x BN, 8 waves (wave32), wave tile 64 x (BN/4).
// Double-buffered LDS; f16 operands staged with global_load_async_to_lds_b128.
// ACT: 0 = relu, 1 = sigmoid.  A_F32: A is fp32 (VGPR-convert while staging).
// ---------------------------------------------------------------------------
template<int BN, int ACT, bool A_F32>
__global__ __launch_bounds__(256) void gemm_bias_act(
    const void* __restrict__ Av, const _Float16* __restrict__ Bt,
    const float* __restrict__ bias, _Float16* __restrict__ Cp,
    int M, int N, int K)
{
    constexpr int BM = 128, BK = 32;
    constexpr int SA = 40, SB = 40;                 // padded LDS strides (halves)
    constexpr int WN = (BN == 128) ? 32 : 16;       // wave tile N
    constexpr int WN_ITER = WN / 16;
    constexpr int WM_ITER = 4;

    __shared__ __align__(16) _Float16 sA[2 * BM * SA];
    __shared__ __align__(16) _Float16 sB[2 * BN * SB];

    const int tid   = threadIdx.x;
    const int lane  = tid & 31;
    const int wid   = tid >> 5;
    const int waveM = wid >> 2;        // 0..1
    const int waveN = wid & 3;         // 0..3
    const int m0 = blockIdx.x * BM;
    const int n0 = blockIdx.y * BN;

    v8f acc[WM_ITER][WN_ITER] = {};

    const int srow  = tid >> 2;         // 0..63
    const int skcol = (tid & 3) * 8;    // 0,8,16,24

    float4 fa[2][2];                    // f32-A staging registers

    auto loadA_f32 = [&](int k0) {
        #pragma unroll
        for (int rr = 0; rr < 2; ++rr) {
            const float* ap = (const float*)Av + (size_t)(m0 + srow + rr * 64) * K + k0 + skcol;
            fa[rr][0] = *(const float4*)(ap);
            fa[rr][1] = *(const float4*)(ap + 4);
        }
    };
    auto commitA_f32 = [&](int buf) {
        #pragma unroll
        for (int rr = 0; rr < 2; ++rr) {
            _Float16* d = &sA[buf * BM * SA + (srow + rr * 64) * SA + skcol];
            d[0] = (_Float16)fa[rr][0].x; d[1] = (_Float16)fa[rr][0].y;
            d[2] = (_Float16)fa[rr][0].z; d[3] = (_Float16)fa[rr][0].w;
            d[4] = (_Float16)fa[rr][1].x; d[5] = (_Float16)fa[rr][1].y;
            d[6] = (_Float16)fa[rr][1].z; d[7] = (_Float16)fa[rr][1].w;
        }
    };
    auto stageA_async = [&](int k0, int buf) {
        #pragma unroll
        for (int rr = 0; rr < 2; ++rr) {
            int r = srow + rr * 64;
            async_copy_b128(lds_off(&sA[buf * BM * SA + r * SA + skcol]),
                            (const _Float16*)Av + (size_t)(m0 + r) * K + k0 + skcol);
        }
    };
    auto stageB_async = [&](int k0, int buf) {
        #pragma unroll
        for (int rr = 0; rr < BN / 64; ++rr) {
            int n = srow + rr * 64;
            async_copy_b128(lds_off(&sB[buf * BN * SB + n * SB + skcol]),
                            Bt + (size_t)(n0 + n) * K + k0 + skcol);
        }
    };

    // ---- prologue: fill buffer 0 ----
    if constexpr (A_F32) { loadA_f32(0); commitA_f32(0); }
    else                 { stageA_async(0, 0); }
    stageB_async(0, 0);
    wait_async0();
    __syncthreads();

    int buf = 0;
    for (int k0 = 0; k0 < K; k0 += BK) {
        const bool last = (k0 + BK >= K);

        // ---- kick off staging of next tile into buf^1 ----
        if (!last) {
            if constexpr (A_F32) loadA_f32(k0 + BK);
            else                 stageA_async(k0 + BK, buf ^ 1);
            stageB_async(k0 + BK, buf ^ 1);
        }

        // ---- fragments per ISA 16-bit layouts, from current buffer ----
        const _Float16* bufA = &sA[buf * BM * SA];
        const _Float16* bufB = &sB[buf * BN * SB];
        const int rA    = lane & 15;
        const int koffA = (lane >> 4) * 8;    // A: lanes 16-31 hold K+8 block
        const int koffB = (lane >> 4) * 8;    // B: lanes 16-31 hold K+8 block

        v16h aF[WM_ITER];
        #pragma unroll
        for (int mi = 0; mi < WM_ITER; ++mi) {
            const _Float16* base = bufA + (waveM * 64 + mi * 16 + rA) * SA;
            v8h lo = *(const v8h*)(base + koffA);
            v8h hi = *(const v8h*)(base + koffA + 16);
            aF[mi] = __builtin_shufflevector(lo, hi, 0,1,2,3,4,5,6,7,8,9,10,11,12,13,14,15);
        }
        v16h bF[WN_ITER];
        #pragma unroll
        for (int ni = 0; ni < WN_ITER; ++ni) {
            const _Float16* base = bufB + (waveN * WN + ni * 16 + rA) * SB + koffB;
            v8h lo = *(const v8h*)(base);
            v8h hi = *(const v8h*)(base + 16);
            bF[ni] = __builtin_shufflevector(lo, hi, 0,1,2,3,4,5,6,7,8,9,10,11,12,13,14,15);
        }

        #pragma unroll
        for (int mi = 0; mi < WM_ITER; ++mi)
            #pragma unroll
            for (int ni = 0; ni < WN_ITER; ++ni)
                acc[mi][ni] = __builtin_amdgcn_wmma_f32_16x16x32_f16(
                    false, aF[mi], false, bF[ni], (short)0, acc[mi][ni], false, false);

        // ---- finish staging of next tile, then rendezvous ----
        if (!last) {
            if constexpr (A_F32) commitA_f32(buf ^ 1);
            wait_async0();
        }
        __syncthreads();
        buf ^= 1;
    }

    // ---- epilogue: bias + activation, store f16 ----
    const int rowHalf = (lane >> 4) * 8;     // C/D layout: lanes 16-31 -> M+8
    const int colLane = lane & 15;
    #pragma unroll
    for (int ni = 0; ni < WN_ITER; ++ni) {
        int col = n0 + waveN * WN + ni * 16 + colLane;
        float bv = bias[col];
        #pragma unroll
        for (int mi = 0; mi < WM_ITER; ++mi) {
            int rbase = m0 + waveM * 64 + mi * 16 + rowHalf;
            #pragma unroll
            for (int j = 0; j < 8; ++j) {
                float v = acc[mi][ni][j] + bv;
                if (ACT == 0) v = v > 0.f ? v : 0.f;
                else          v = 1.f / (1.f + __expf(-v));
                Cp[(size_t)(rbase + j) * N + col] = (_Float16)v;
            }
        }
    }
}

// ---------------------------------------------------------------------------
// Per-bag head: d = G2@D3+db3, gumbel softmax, argsort-dropout keep-top-20,
// masked re-softmax, out[n] = sum_c w[c] * dot(H3[n,c,:], E) + eb.
// One block of 64 threads per bag (c = threadIdx.x).
// ---------------------------------------------------------------------------
__global__ __launch_bounds__(64) void bag_head(
    const _Float16* __restrict__ H3, const _Float16* __restrict__ G2,
    const float* __restrict__ D3, const float* __restrict__ db3,
    const float* __restrict__ E,  const float* __restrict__ eb,
    const float* __restrict__ m,  const float* __restrict__ u,
    float* __restrict__ w_out, float* __restrict__ out)
{
    const int n = blockIdx.x;
    const int c = threadIdx.x;              // 0..63
    const size_t row = (size_t)n * 64 + c;

    // d-value: dot(G2 row (64), D3) + db3
    float dval = db3[0];
    const _Float16* g2 = G2 + row * 64;
    #pragma unroll 8
    for (int j = 0; j < 64; ++j) dval += (float)g2[j] * D3[j];

    // he = dot(H3 row (128), E)
    float he = 0.f;
    const _Float16* h3 = H3 + row * 128;
    #pragma unroll 8
    for (int j = 0; j < 128; ++j) he += (float)h3[j] * E[j];

    float logit = m[row] * dval;
    float gum   = -__logf(-__logf(u[row]));
    float s     = (logit + gum) / TAU;

    __shared__ float sb[64], wb[64], red[64], hb[64];
    __shared__ int   kb[64];
    sb[c] = s; hb[c] = he;
    __syncthreads();

    float mx = -1e30f;
    for (int j = 0; j < 64; ++j) mx = fmaxf(mx, sb[j]);
    float e = __expf(s - mx);
    red[c] = e;
    __syncthreads();
    float sum = 0.f;
    for (int j = 0; j < 64; ++j) sum += red[j];
    float w0 = e / sum;
    wb[c] = w0;
    __syncthreads();

    // descending rank, ties broken so larger index is kept first
    // (matches stable ascending argsort dropping first 44)
    int r = 0;
    for (int j = 0; j < 64; ++j) {
        float wj = wb[j];
        r += (wj > w0) || (wj == w0 && j > c);
    }
    int keep = (r < KEEP) ? 1 : 0;
    kb[c] = keep;
    __syncthreads();

    // masked softmax over kept w0 values
    float mk = -1e30f;
    for (int j = 0; j < 64; ++j) if (kb[j]) mk = fmaxf(mk, wb[j]);
    float ez = keep ? __expf(w0 - mk) : 0.f;
    __syncthreads();           // red[] reuse
    red[c] = ez;
    __syncthreads();
    float s2 = 0.f;
    for (int j = 0; j < 64; ++j) s2 += red[j];
    float wf = ez / s2;

    w_out[row] = wf;

    __syncthreads();           // red[] reuse
    red[c] = wf * hb[c];
    __syncthreads();
    if (c == 0) {
        float o = eb[0];
        for (int j = 0; j < 64; ++j) o += red[j];
        out[n] = o;
    }
}

// ---------------------------------------------------------------------------
extern "C" void kernel_launch(void* const* d_in, const int* in_sizes, int n_in,
                              void* d_out, int out_size, void* d_ws, size_t ws_size,
                              hipStream_t stream) {
    // inputs (setup_inputs order)
    const float* x   = (const float*)d_in[0];
    const float* m   = (const float*)d_in[1];
    const float* u   = (const float*)d_in[2];
    const float* W1  = (const float*)d_in[3];  const float* b1  = (const float*)d_in[4];
    const float* W2  = (const float*)d_in[5];  const float* b2  = (const float*)d_in[6];
    const float* W3  = (const float*)d_in[7];  const float* b3  = (const float*)d_in[8];
    const float* D1  = (const float*)d_in[9];  const float* db1 = (const float*)d_in[10];
    const float* D2  = (const float*)d_in[11]; const float* db2 = (const float*)d_in[12];
    const float* D3  = (const float*)d_in[13]; const float* db3 = (const float*)d_in[14];
    const float* E   = (const float*)d_in[15]; const float* eb  = (const float*)d_in[16];

    const int M = 2048 * 64;   // 131072 rows

    // workspace carve-up (bytes)
    char* ws = (char*)d_ws;
    size_t off = 0;
    auto alloc = [&](size_t bytes) { char* p = ws + off; off = (off + bytes + 255) & ~(size_t)255; return p; };
    _Float16* W1t = (_Float16*)alloc(256 * 512 * 2);
    _Float16* W2t = (_Float16*)alloc(256 * 256 * 2);
    _Float16* W3t = (_Float16*)alloc(128 * 256 * 2);
    _Float16* D1t = (_Float16*)alloc(128 * 128 * 2);
    _Float16* D2t = (_Float16*)alloc(64 * 128 * 2);
    _Float16* H1  = (_Float16*)alloc((size_t)M * 256 * 2);
    _Float16* H2  = (_Float16*)alloc((size_t)M * 256 * 2);
    _Float16* H3  = (_Float16*)alloc((size_t)M * 128 * 2);
    _Float16* G1  = H1;   // alias: H1 dead after GEMM2
    _Float16* G2  = H2;   // alias: H2 dead after GEMM3

    // weight transpose/convert
    convert_wt<<<(512 * 256 + 255) / 256, 256, 0, stream>>>(W1, W1t, 512, 256);
    convert_wt<<<(256 * 256 + 255) / 256, 256, 0, stream>>>(W2, W2t, 256, 256);
    convert_wt<<<(256 * 128 + 255) / 256, 256, 0, stream>>>(W3, W3t, 256, 128);
    convert_wt<<<(128 * 128 + 255) / 256, 256, 0, stream>>>(D1, D1t, 128, 128);
    convert_wt<<<(128 * 64  + 255) / 256, 256, 0, stream>>>(D2, D2t, 128, 64);

    // GEMM chain
    gemm_bias_act<128, 0, true ><<<dim3(M / 128, 2), 256, 0, stream>>>(x,  W1t, b1,  H1, M, 256, 512);
    gemm_bias_act<128, 0, false><<<dim3(M / 128, 2), 256, 0, stream>>>(H1, W2t, b2,  H2, M, 256, 256);
    gemm_bias_act<128, 0, false><<<dim3(M / 128, 1), 256, 0, stream>>>(H2, W3t, b3,  H3, M, 128, 256);
    gemm_bias_act<128, 1, false><<<dim3(M / 128, 1), 256, 0, stream>>>(H3, D1t, db1, G1, M, 128, 128);
    gemm_bias_act<64,  1, false><<<dim3(M / 128, 1), 256, 0, stream>>>(G1, D2t, db2, G2, M, 64,  128);

    // head: w (2048*64 floats) then out (2048 floats) in d_out
    float* w_out = (float*)d_out;
    float* out   = (float*)d_out + 2048 * 64;
    bag_head<<<2048, 64, 0, stream>>>(H3, G2, D3, db3, E, eb, m, u, w_out, out);
}
